// DGCNRegressionModule_30021821399850
// MI455X (gfx1250) — matrix operand order repaired and checked by reading, not compiled, not run on hardware
//
#include <hip/hip_runtime.h>
#include <hip/hip_bf16.h>

// ---------------------------------------------------------------------------
// DGCNN regression, MI455X (gfx1250, wave32, WMMA bf16 16x16x32, f32 accum)
// B=4 clouds, N=2048 pts, K=20 knn, H=512, NC=512
// ---------------------------------------------------------------------------

typedef __bf16 bf16_t;
typedef __attribute__((ext_vector_type(16))) __bf16 v16bf;
typedef __attribute__((ext_vector_type(8)))  float  v8f;
typedef __attribute__((ext_vector_type(4)))  unsigned int v4u;

#define NPTS   2048
#define NB     4
#define HD     512
#define KNN    20
#define NEGINF (-3.402823466e38f)

// ---- WMMA operand loaders (contiguous k = 16h+s, same order in A and B) ----------------
// A-matrix 16x32 bf16: lane L holds row M=L&15; lanes<16: K 0..15,
// lanes>=16: K 16..31. rp = row pointer at current k-chunk.
__device__ __forceinline__ v16bf load_a32(const bf16_t* rp, int hi) {
  union { v16bf v; v4u q[2]; } u;
  u.q[0] = *(const v4u*)(rp + (hi ? 16 : 0));
  u.q[1] = *(const v4u*)(rp + (hi ? 24 : 8));
  return u.v;
}
// B-matrix 32x16 bf16: lane L holds col N=L&15; K = (L<16 ? 0..15 : 16..31),
// 16 contiguous elements. cp already offset by the K half-select.
__device__ __forceinline__ v16bf load_b32(const bf16_t* cp) {
  union { v16bf v; v4u q[2]; } u;
  u.q[0] = *(const v4u*)(cp);
  u.q[1] = *(const v4u*)(cp + 8);
  return u.v;
}
__device__ __forceinline__ v8f wmma_bf16(v16bf a, v16bf b, v8f c) {
  return __builtin_amdgcn_wmma_f32_16x16x32_bf16(false, a, false, b,
                                                 (short)0, c, false, false);
}
__device__ __forceinline__ float elu(float x) {
  return x > 0.f ? x : (__expf(x) - 1.f);
}

// ---------------------------------------------------------------------------
// Weight prep: fp32 [din,dout] -> bf16 transposed [dout,din] (cols contiguous)
// ---------------------------------------------------------------------------
__global__ void k_w2bf(const float* __restrict__ W, bf16_t* __restrict__ Wt,
                       int din, int dout) {
  int gid = blockIdx.x * 256 + threadIdx.x;
  if (gid >= din * dout) return;
  int o = gid / din, i = gid - o * din;
  Wt[gid] = (bf16_t)W[i * dout + o];
}

// ---------------------------------------------------------------------------
// ffm: x[8192,3] @ W[3,512] + b -> xb (f32 + bf16 mirror)
// ---------------------------------------------------------------------------
__global__ void k_ffm(const float* __restrict__ x, const float* __restrict__ W,
                      const float* __restrict__ bias,
                      float* __restrict__ xb, bf16_t* __restrict__ xb16) {
  int gid = blockIdx.x * 256 + threadIdx.x;        // 8192*512
  int p = gid >> 9, h = gid & 511;
  float v = bias[h] + x[p*3+0]*W[h] + x[p*3+1]*W[512+h] + x[p*3+2]*W[1024+h];
  xb[gid] = v;
  xb16[gid] = (bf16_t)v;
}

// ---------------------------------------------------------------------------
// squared norms: one wave per point
// ---------------------------------------------------------------------------
__global__ __launch_bounds__(256) void k_sq(const float* __restrict__ xb,
                                            float* __restrict__ sq) {
  int wave = threadIdx.x >> 5, lane = threadIdx.x & 31;
  int p = blockIdx.x * 8 + wave;
  const float* r = xb + (size_t)p * HD;
  float s = 0.f;
  for (int i = 0; i < 16; i++) { float v = r[lane + i*32]; s += v*v; }
  for (int off = 16; off; off >>= 1) s += __shfl_xor(s, off, 32);
  if (lane == 0) sq[p] = s;
}

// ---------------------------------------------------------------------------
// kNN: per workgroup, 16 query rows of one cloud. Phase 1: distances for all
// 2048 candidates via WMMA Gram tiles into LDS. Phase 2: iterative top-20.
// ---------------------------------------------------------------------------
__global__ __launch_bounds__(256) void k_knn(const bf16_t* __restrict__ xb16,
                                             const float* __restrict__ sq,
                                             int* __restrict__ knn) {
  __shared__ __align__(16) bf16_t Apan[16 * HD];      // 16 KB
  __shared__ __align__(16) float  D[16 * NPTS];       // 128 KB
  int wg  = blockIdx.x;                // 512 = 4 clouds * 128 row tiles
  int b   = wg >> 7;
  int rt  = wg & 127;
  int row0 = b * NPTS + rt * 16;       // global row base
  int tid = threadIdx.x;
  // stage the 16x512 bf16 query panel
  {
    const v4u* src = (const v4u*)(xb16 + (size_t)row0 * HD);
    v4u* dst = (v4u*)Apan;
    for (int i = tid; i < 16 * HD / 8; i += 256) dst[i] = src[i];
  }
  __syncthreads();
  int wave = tid >> 5, lane = tid & 31;
  int rowA = lane & 15, hi = lane >> 4;
  const bf16_t* arow = Apan + rowA * HD;
  float sqr[8];
  for (int v = 0; v < 8; v++) sqr[v] = sq[row0 + v + (hi ? 8 : 0)];
  // 128 column tiles across 8 waves
  for (int t = wave; t < NPTS / 16; t += 8) {
    int colg = b * NPTS + t * 16 + (lane & 15);
    const bf16_t* bcol = xb16 + (size_t)colg * HD + (hi ? 16 : 0);
    v8f acc = {};
    for (int kk = 0; kk < HD; kk += 32) {
      __builtin_prefetch(bcol + kk + 128, 0, 1);
      v16bf a  = load_a32(arow + kk, hi);
      v16bf bb = load_b32(bcol + kk);
      acc = wmma_bf16(a, bb, acc);
    }
    float sqc = sq[colg];
    for (int v = 0; v < 8; v++) {
      int r = v + (hi ? 8 : 0);
      D[r * NPTS + t * 16 + (lane & 15)] = sqr[v] + sqc - 2.f * acc[v];
    }
  }
  __syncthreads();
  // top-20 per row: 8 waves x 2 rows; repeated wave-min with taken bitmask
  for (int rr = 0; rr < 2; rr++) {
    int r = wave * 2 + rr;
    unsigned long long taken = 0ull;
    for (int it = 0; it < KNN; it++) {
      unsigned bestKey = 0xFFFFFFFFu;
      int bestCol = 0x7FFFFFFF;
      for (int jj = 0; jj < NPTS / 32; jj++) {
        if (taken & (1ull << jj)) continue;
        int col = jj * 32 + lane;
        unsigned u = __float_as_uint(D[r * NPTS + col]);
        u = (u & 0x80000000u) ? ~u : (u | 0x80000000u);   // sortable key
        if (u < bestKey) { bestKey = u; bestCol = col; }
      }
      for (int off = 16; off; off >>= 1) {
        unsigned ok = __shfl_xor(bestKey, off, 32);
        int      oc = __shfl_xor(bestCol, off, 32);
        if (ok < bestKey || (ok == bestKey && oc < bestCol)) {
          bestKey = ok; bestCol = oc;
        }
      }
      if ((bestCol & 31) == lane) taken |= 1ull << (bestCol >> 5);
      if (lane == 0) knn[(size_t)(row0 + r) * KNN + it] = bestCol; // cloud-local
    }
  }
}

// ---------------------------------------------------------------------------
// One WMMA GEMM stage of the edge MLP: [16 x K] @ Wt[512 x K]^T -> 16 x 512.
// Each of 8 waves owns 4 N-tiles of 16 output columns.
// ---------------------------------------------------------------------------
__device__ __forceinline__ void gemm_stage(const bf16_t* __restrict__ A, int K,
                                           const bf16_t* __restrict__ Wt,
                                           const float* __restrict__ bias,
                                           bf16_t* __restrict__ out_bf,  // ELU+bf16 path
                                           float*  __restrict__ out_f) { // f32 path
  int tid = threadIdx.x, wave = tid >> 5, lane = tid & 31;
  int rowA = lane & 15, hi = lane >> 4;
  const bf16_t* arow = A + rowA * K;
  for (int nti = 0; nti < 4; nti++) {
    int col = (wave * 4 + nti) * 16 + (lane & 15);
    const bf16_t* bcol = Wt + (size_t)col * K + (hi ? 16 : 0);
    v8f acc = {};
    for (int kk = 0; kk < K; kk += 32) {
      __builtin_prefetch(bcol + kk + 128, 0, 1);
      v16bf a  = load_a32(arow + kk, hi);
      v16bf bb = load_b32(bcol + kk);
      acc = wmma_bf16(a, bb, acc);
    }
    float bv = bias[col];
    for (int v = 0; v < 8; v++) {
      int r = v + (hi ? 8 : 0);
      float o = acc[v] + bv;
      if (out_bf) out_bf[r * HD + col] = (bf16_t)elu(o);
      else        out_f [r * HD + col] = o;
    }
  }
}

// ---------------------------------------------------------------------------
// EdgeConv: workgroup = 8 points = 160 edges = 10 M-tiles of 16.
// e = [xi, xj-xi] (1024) -> 512 -> 512 -> 512, max over k per point.
// ---------------------------------------------------------------------------
__global__ __launch_bounds__(256) void k_edgeconv(
    const float* __restrict__ xb, const int* __restrict__ knn,
    const bf16_t* __restrict__ W0t, const float* __restrict__ b0,
    const bf16_t* __restrict__ W1t, const float* __restrict__ b1,
    const bf16_t* __restrict__ W2t, const float* __restrict__ b2,
    float* __restrict__ y) {
  __shared__ __align__(16) bf16_t e_tile[16 * 1024];  // 32 KB
  __shared__ __align__(16) bf16_t hA[16 * HD];        // 16 KB
  __shared__ __align__(16) bf16_t hB[16 * HD];        // 16 KB
  __shared__ __align__(16) float  h3[16 * HD];        // 32 KB
  __shared__ float yacc[8 * HD];                      // 16 KB
  int wg = blockIdx.x;                 // 1024 (= 8192/8)
  int p0 = wg * 8;
  int cloudBase = (p0 / NPTS) * NPTS;
  int tid = threadIdx.x;
  for (int i = tid; i < 8 * HD; i += 256) yacc[i] = NEGINF;
  for (int mt = 0; mt < 10; mt++) {
    __syncthreads();
    // build the 16x1024 bf16 edge-feature tile
    for (int i = tid; i < 16 * 1024; i += 256) {
      int r = i >> 10, c = i & 1023;
      int le = mt * 16 + r;                 // 0..159
      int pl = le / KNN, k = le - pl * KNN;
      int p = p0 + pl;
      int q = cloudBase + knn[(size_t)p * KNN + k];
      int cc = c & 511;
      float xi = xb[(size_t)p * HD + cc];
      float val = (c < HD) ? xi : (xb[(size_t)q * HD + cc] - xi);
      e_tile[i] = (bf16_t)val;
    }
    __syncthreads();
    gemm_stage(e_tile, 1024, W0t, b0, hA, nullptr);   // ELU
    __syncthreads();
    gemm_stage(hA, 512, W1t, b1, hB, nullptr);        // ELU
    __syncthreads();
    gemm_stage(hB, 512, W2t, b2, nullptr, h3);        // linear, f32
    __syncthreads();
    // max-fold over k into per-point accumulator (col-parallel, race-free)
    for (int c = tid; c < HD; c += 256) {
      for (int r = 0; r < 16; r++) {
        int pl = (mt * 16 + r) / KNN;
        float v = h3[r * HD + c];
        float cur = yacc[pl * HD + c];
        yacc[pl * HD + c] = v > cur ? v : cur;
      }
    }
  }
  __syncthreads();
  for (int i = tid; i < 8 * HD; i += 256) y[(size_t)p0 * HD + i] = yacc[i];
}

// ---------------------------------------------------------------------------
// LayerNorm + ELU -> bf16 (one wave per point)
// ---------------------------------------------------------------------------
__global__ __launch_bounds__(256) void k_ln_elu(const float* __restrict__ y,
                                                const float* __restrict__ g,
                                                const float* __restrict__ be,
                                                bf16_t* __restrict__ z) {
  int wave = threadIdx.x >> 5, lane = threadIdx.x & 31;
  int p = blockIdx.x * 8 + wave;
  const float* yr = y + (size_t)p * HD;
  float vals[16], s = 0.f, s2 = 0.f;
  for (int i = 0; i < 16; i++) { vals[i] = yr[lane + i*32]; s += vals[i]; }
  for (int off = 16; off; off >>= 1) s += __shfl_xor(s, off, 32);
  float mu = s * (1.f / HD);
  for (int i = 0; i < 16; i++) { float d = vals[i] - mu; s2 += d * d; }
  for (int off = 16; off; off >>= 1) s2 += __shfl_xor(s2, off, 32);
  float rstd = rsqrtf(s2 * (1.f / HD) + 1e-5f);
  for (int i = 0; i < 16; i++) {
    int c = lane + i * 32;
    float v = (vals[i] - mu) * rstd * g[c] + be[c];
    z[(size_t)p * HD + c] = (bf16_t)elu(v);
  }
}

// ---------------------------------------------------------------------------
// WMMA GEMM [8192 x 512] @ Wt^T + bias; optional alpha-residual into xb.
// Workgroup = one 16-row M-tile; 8 waves x 4 N-tiles.
// ---------------------------------------------------------------------------
__global__ __launch_bounds__(256) void k_out(
    const bf16_t* __restrict__ Zin, const bf16_t* __restrict__ Wt,
    const float* __restrict__ bias,
    const float* __restrict__ xb_in, const float* __restrict__ alpha,
    float* __restrict__ out_f32, bf16_t* __restrict__ out_b16) {
  int mt = blockIdx.x;                 // 512 tiles
  int tid = threadIdx.x, wave = tid >> 5, lane = tid & 31;
  int rowA = lane & 15, hi = lane >> 4;
  const bf16_t* arow = Zin + (size_t)(mt * 16 + rowA) * HD;
  float al = alpha ? alpha[0] : 0.f;
  for (int nti = 0; nti < 4; nti++) {
    int col = (wave * 4 + nti) * 16 + (lane & 15);
    const bf16_t* bcol = Wt + (size_t)col * HD + (hi ? 16 : 0);
    v8f acc = {};
    for (int kk = 0; kk < HD; kk += 32) {
      __builtin_prefetch(bcol + kk + 128, 0, 1);
      acc = wmma_bf16(load_a32(arow + kk, hi), load_b32(bcol + kk), acc);
    }
    float bv = bias[col];
    for (int v = 0; v < 8; v++) {
      int r = v + (hi ? 8 : 0);
      size_t o = (size_t)(mt * 16 + r) * HD + col;
      float val = acc[v] + bv;
      if (alpha) val = xb_in[o] + al * val;
      out_f32[o] = val;
      if (out_b16) out_b16[o] = (bf16_t)val;
    }
  }
}

// ---------------------------------------------------------------------------
// global max pool over N per cloud
// ---------------------------------------------------------------------------
__global__ void k_gmax(const float* __restrict__ r, float* __restrict__ gmax) {
  int gid = blockIdx.x * 256 + threadIdx.x;      // 4*512
  int b = gid >> 9, h = gid & 511;
  const float* base = r + (size_t)b * NPTS * HD + h;
  float m = NEGINF;
  for (int p = 0; p < NPTS; p++) m = fmaxf(m, base[(size_t)p * HD]);
  gmax[gid] = m;
}

// ---------------------------------------------------------------------------
// tiny regression head: [4,512] -> elu 256 -> elu 128 -> 512
// ---------------------------------------------------------------------------
__global__ __launch_bounds__(512) void k_head(
    const float* __restrict__ gmax,
    const float* __restrict__ W1, const float* __restrict__ b1,
    const float* __restrict__ W2, const float* __restrict__ b2,
    const float* __restrict__ W3, const float* __restrict__ b3,
    float* __restrict__ out) {
  __shared__ float h1[4 * 256];
  __shared__ float h2[4 * 128];
  int tid = threadIdx.x;
  for (int i = tid; i < 4 * 256; i += 512) {
    int row = i >> 8, j = i & 255;
    float s = b1[j];
    for (int k = 0; k < 512; k++) s += gmax[row * 512 + k] * W1[k * 256 + j];
    h1[i] = elu(s);
  }
  __syncthreads();
  for (int i = tid; i < 4 * 128; i += 512) {
    int row = i >> 7, j = i & 127;
    float s = b2[j];
    for (int k = 0; k < 256; k++) s += h1[row * 256 + k] * W2[k * 128 + j];
    h2[i] = elu(s);
  }
  __syncthreads();
  for (int i = tid; i < 4 * 512; i += 512) {
    int row = i >> 9, j = i & 511;
    float s = b3[j];
    for (int k = 0; k < 128; k++) s += h2[row * 128 + k] * W3[k * 512 + j];
    out[i] = s;
  }
}

// ---------------------------------------------------------------------------
// Orchestration. Input order (insertion order of setup_inputs pytree):
// 0:x 1:batch 2:ffmW 3:ffmB; per layer l (base 4+11l):
//   W0,b0,W1,b1,W2,b2, ln_g,ln_b, Wout,bout, alpha
// reg (base 37): Wr0,br0, Wr1,br1, Wr2,br2, Wr3,br3
// ---------------------------------------------------------------------------
extern "C" void kernel_launch(void* const* d_in, const int* in_sizes, int n_in,
                              void* d_out, int out_size, void* d_ws, size_t ws_size,
                              hipStream_t stream) {
  (void)in_sizes; (void)n_in; (void)out_size; (void)ws_size;
  const float* x    = (const float*)d_in[0];
  const float* ffmW = (const float*)d_in[2];
  const float* ffmB = (const float*)d_in[3];
  auto LIN = [](int l, int j) { return 4 + l * 11 + j; };

  char* base = (char*)d_ws;
  size_t off = 0;
  auto alloc = [&](size_t bytes) -> char* {
    off = (off + 255) & ~(size_t)255;
    char* p = base + off; off += bytes; return p;
  };
  const size_t NP = (size_t)NB * NPTS;            // 8192
  float*  xb   = (float*) alloc(NP * HD * 4);
  bf16_t* xb16 = (bf16_t*)alloc(NP * HD * 2);
  float*  y    = (float*) alloc(NP * HD * 4);
  bf16_t* z16  = (bf16_t*)alloc(NP * HD * 2);
  float*  rbuf = (float*) alloc(NP * HD * 4);
  float*  sq   = (float*) alloc(NP * 4);
  int*    knn  = (int*)   alloc(NP * KNN * 4);
  float*  gmx  = (float*) alloc(NB * HD * 4);
  bf16_t *W0t[3], *W1t[3], *W2t[3], *Wot[3];
  for (int l = 0; l < 3; l++) {
    W0t[l] = (bf16_t*)alloc((size_t)512 * 1024 * 2);
    W1t[l] = (bf16_t*)alloc((size_t)512 * 512 * 2);
    W2t[l] = (bf16_t*)alloc((size_t)512 * 512 * 2);
    Wot[l] = (bf16_t*)alloc((size_t)512 * 512 * 2);
  }
  bf16_t* Wr0t = (bf16_t*)alloc((size_t)512 * 512 * 2);

  // weight prep (bf16, transposed so GEMM B-columns are contiguous)
  for (int l = 0; l < 3; l++) {
    k_w2bf<<<(1024*512+255)/256, 256, 0, stream>>>((const float*)d_in[LIN(l,0)], W0t[l], 1024, 512);
    k_w2bf<<<( 512*512+255)/256, 256, 0, stream>>>((const float*)d_in[LIN(l,2)], W1t[l],  512, 512);
    k_w2bf<<<( 512*512+255)/256, 256, 0, stream>>>((const float*)d_in[LIN(l,4)], W2t[l],  512, 512);
    k_w2bf<<<( 512*512+255)/256, 256, 0, stream>>>((const float*)d_in[LIN(l,8)], Wot[l],  512, 512);
  }
  k_w2bf<<<(512*512+255)/256, 256, 0, stream>>>((const float*)d_in[37], Wr0t, 512, 512);

  k_ffm<<<NP * HD / 256, 256, 0, stream>>>(x, ffmW, ffmB, xb, xb16);

  for (int l = 0; l < 3; l++) {
    k_sq <<<NP / 8, 256, 0, stream>>>(xb, sq);
    k_knn<<<NP / 16, 256, 0, stream>>>(xb16, sq, knn);
    k_edgeconv<<<NP / 8, 256, 0, stream>>>(
        xb, knn,
        W0t[l], (const float*)d_in[LIN(l,1)],
        W1t[l], (const float*)d_in[LIN(l,3)],
        W2t[l], (const float*)d_in[LIN(l,5)], y);
    k_ln_elu<<<NP / 8, 256, 0, stream>>>(
        y, (const float*)d_in[LIN(l,6)], (const float*)d_in[LIN(l,7)], z16);
    k_out<<<NP / 16, 256, 0, stream>>>(
        z16, Wot[l], (const float*)d_in[LIN(l,9)],
        xb, (const float*)d_in[LIN(l,10)], xb, xb16);   // xb += alpha*(...)
  }
  // reg[0] linear (plain GEMM), then global max pool + head
  k_out<<<NP / 16, 256, 0, stream>>>(xb16, Wr0t, (const float*)d_in[38],
                                     nullptr, nullptr, rbuf, nullptr);
  k_gmax<<<NB * HD / 256, 256, 0, stream>>>(rbuf, gmx);
  k_head<<<1, 512, 0, stream>>>(gmx,
      (const float*)d_in[39], (const float*)d_in[40],
      (const float*)d_in[41], (const float*)d_in[42],
      (const float*)d_in[43], (const float*)d_in[44], (float*)d_out);
}
